// AnyprecisionLinear_39213051412735
// MI455X (gfx1250) — matrix operand order repaired and checked by reading, not compiled, not run on hardware
//
#include <hip/hip_runtime.h>

typedef __attribute__((ext_vector_type(16))) _Float16 v16h;
typedef __attribute__((ext_vector_type(8)))  float    v8f;

#define IN_FEATURES  4096
#define OUT_FEATURES 11008
#define BATCH        4096

#define M_TILE  256
#define N_TILE  128
#define KC      32
#define K_TILES (IN_FEATURES / KC)   // 128
#define LDSTR   40                   // padded K-stride in halves (80B: banks spread, 16B aligned)
#define XBUF    (M_TILE * LDSTR)     // halves per x buffer (10240)
#define WBUF    (N_TILE * LDSTR)     // halves per w buffer (5120)

union Frag16 { v16h h; uint4 q[2]; };

// async global->LDS (ASYNCcnt path); vdst = per-lane LDS byte address, vaddr = 64-bit global
#define ASYNC_X_B128(ldsaddr, gptr)                                   \
    asm volatile("global_load_async_to_lds_b128 %0, %1, off"          \
                 :: "v"(ldsaddr), "v"(gptr) : "memory")

// raw split barrier: drain only DScnt (our ds stores + frag reads of [cur])
#define BAR_DS()                                                      \
    asm volatile("s_wait_dscnt 0x0\n\t"                               \
                 "s_barrier_signal -1\n\t"                            \
                 "s_barrier_wait -1" ::: "memory")

// raw split barrier draining ASYNCcnt + DScnt; global loads stay outstanding
#define BAR_ASYNC_DS()                                                \
    asm volatile("s_wait_asynccnt 0x0\n\t"                            \
                 "s_wait_dscnt 0x0\n\t"                               \
                 "s_barrier_signal -1\n\t"                            \
                 "s_barrier_wait -1" ::: "memory")

__global__ __launch_bounds__(256)
void lutgemm_wmma_kernel(const _Float16* __restrict__ x,
                         const _Float16* __restrict__ lut,
                         const int*      __restrict__ qw,
                         float*          __restrict__ y)
{
    __shared__ _Float16 xs[2 * XBUF];       // 40 KB  A tiles (double buffered, async-filled)
    __shared__ _Float16 ws[2 * WBUF];       // 20 KB  dequantized W tiles (double buffered)
    __shared__ _Float16 luts[N_TILE * 16];  //  4 KB  per-channel LUT rows for this N tile

    const int tid  = threadIdx.x;
    const int lane = tid & 31;
    const int wave = tid >> 5;          // 0..7
    const int wm   = wave & 3;          // M quarter (64 rows)
    const int wn   = wave >> 2;         // N half    (64 cols)
    const int l16  = lane & 15;
    const int lhi  = lane >> 4;         // 0/1: lane half

    const int m0 = blockIdx.x * M_TILE; // blockIdx.x fastest => M sweep reuses qweight tile in L2
    const int n0 = blockIdx.y * N_TILE;

    // ---- stage LUT rows for this N tile: 128 rows x 16 halves = 256 uint4 ----
    ((uint4*)luts)[tid] = ((const uint4*)(lut + (size_t)n0 * 16))[tid];

    // ---- per-thread fetch mapping ----
    // x: vector id v = tid + 256*j (j=0..3): row = (tid>>2) + 64*j, seg = tid&3 (8 halves)
    // q: segment id s = tid + 256*j (j=0..1): n   = (tid>>2) + 64*j, seg = tid&3 (8 ints)
    const int r0 = tid >> 2, s0 = tid & 3;
    const _Float16* xg[4];
    unsigned        xls[4];
#pragma unroll
    for (int j = 0; j < 4; ++j) {
        xg[j]  = x + (size_t)(m0 + r0 + 64 * j) * IN_FEATURES + s0 * 8;
        xls[j] = (unsigned)(uintptr_t)&xs[(r0 + 64 * j) * LDSTR + s0 * 8];
    }
    const int* qg0 = qw + (size_t)(n0 + r0)      * IN_FEATURES + s0 * 8;
    const int* qg1 = qw + (size_t)(n0 + r0 + 64) * IN_FEATURES + s0 * 8;
    const unsigned xbufbytes = XBUF * 2;

    // dequantize 8 codes through this thread's LDS LUT row; codes guaranteed in [0,16)
    auto dequant_store = [&](int buf, const int4& qa, const int4& qb, int rr) {
        union { _Float16 h[8]; uint4 v; } w;
        const _Float16* l = &luts[rr * 16];
        w.h[0] = l[qa.x]; w.h[1] = l[qa.y];
        w.h[2] = l[qa.z]; w.h[3] = l[qa.w];
        w.h[4] = l[qb.x]; w.h[5] = l[qb.y];
        w.h[6] = l[qb.z]; w.h[7] = l[qb.w];
        *(uint4*)&ws[buf * WBUF + rr * LDSTR + s0 * 8] = w.v;
    };

    v8f acc[4][4];
    const v8f vzero = {0.f, 0.f, 0.f, 0.f, 0.f, 0.f, 0.f, 0.f};
#pragma unroll
    for (int mi = 0; mi < 4; ++mi)
#pragma unroll
        for (int ni = 0; ni < 4; ++ni) acc[mi][ni] = vzero;

    // ---- prologue: async x(0) -> xs[0]; q(0) -> regs ----
#pragma unroll
    for (int j = 0; j < 4; ++j) ASYNC_X_B128(xls[j], xg[j]);
    int4 qa0 = ((const int4*)qg0)[0], qb0 = ((const int4*)qg0)[1];
    int4 qa1 = ((const int4*)qg1)[0], qb1 = ((const int4*)qg1)[1];

    BAR_DS();                               // luts visible to all waves

    dequant_store(0, qa0, qb0, r0);         // ws[0] <- tile 0
    dequant_store(0, qa1, qb1, r0 + 64);
    qa0 = ((const int4*)(qg0 + KC))[0]; qb0 = ((const int4*)(qg0 + KC))[1];   // q(1)
    qa1 = ((const int4*)(qg1 + KC))[0]; qb1 = ((const int4*)(qg1 + KC))[1];

    BAR_ASYNC_DS();                         // tiles[0] ready everywhere

    for (int kt = 0; kt < K_TILES; ++kt) {
        const int cur = kt & 1;
        const int nxt = cur ^ 1;

        if (kt + 1 < K_TILES) {
            // dequant q(kt+1) (already in regs) into ws[nxt]
            dequant_store(nxt, qa0, qb0, r0);
            dequant_store(nxt, qa1, qb1, r0 + 64);
            // async x(kt+1) -> xs[nxt]
            const size_t ko = (size_t)(kt + 1) * KC;
#pragma unroll
            for (int j = 0; j < 4; ++j)
                ASYNC_X_B128(xls[j] + nxt * xbufbytes, xg[j] + ko);
            // q(kt+2) -> regs; wait lands at next iteration's dequant, not at the barrier
            if (kt + 2 < K_TILES) {
                const size_t ko2 = (size_t)(kt + 2) * KC;
                qa0 = ((const int4*)(qg0 + ko2))[0]; qb0 = ((const int4*)(qg0 + ko2))[1];
                qa1 = ((const int4*)(qg1 + ko2))[0]; qb1 = ((const int4*)(qg1 + ko2))[1];
            }
        }

        // ---- fragments per ISA VGPR layouts, from current buffers ----
        Frag16 a[4];
        const int akoff = lhi * 8;     // A: lanes 0-15 -> K0..7 & K16..23; lanes 16-31 -> +8
#pragma unroll
        for (int mi = 0; mi < 4; ++mi) {
            const _Float16* p = &xs[cur * XBUF + (wm * 64 + mi * 16 + l16) * LDSTR + akoff];
            a[mi].q[0] = *(const uint4*)p;
            a[mi].q[1] = *(const uint4*)(p + 16);
        }
        const int bkoff = lhi * 8;     // B: same K order as A (K0..7 & K16..23; +8 for lanes 16-31)
#pragma unroll
        for (int ni = 0; ni < 4; ++ni) {
            Frag16 b;
            const _Float16* p = &ws[cur * WBUF + (wn * 64 + ni * 16 + l16) * LDSTR + bkoff];
            b.q[0] = *(const uint4*)p;
            b.q[1] = *(const uint4*)(p + 16);
#pragma unroll
            for (int mi = 0; mi < 4; ++mi)
                acc[mi][ni] = __builtin_amdgcn_wmma_f32_16x16x32_f16(
                    false, a[mi].h, false, b.h,
                    (short)0, acc[mi][ni], false, false);
        }

        BAR_ASYNC_DS();   // [cur] reads + ws[nxt] stores drained; async x(kt+1) landed
    }

    // ---- epilogue: C layout VGPR r -> M = r + 8*laneHalf; lane&15 -> N. NT stores. ----
#pragma unroll
    for (int mi = 0; mi < 4; ++mi) {
#pragma unroll
        for (int ni = 0; ni < 4; ++ni) {
            const int row0 = m0 + wm * 64 + mi * 16 + lhi * 8;
            const int col  = n0 + wn * 64 + ni * 16 + l16;
            float* yp = y + (size_t)row0 * OUT_FEATURES + col;
#pragma unroll
            for (int r = 0; r < 8; ++r)
                __builtin_nontemporal_store(acc[mi][ni][r], yp + (size_t)r * OUT_FEATURES);
        }
    }
}

extern "C" void kernel_launch(void* const* d_in, const int* in_sizes, int n_in,
                              void* d_out, int out_size, void* d_ws, size_t ws_size,
                              hipStream_t stream) {
    const _Float16* x   = (const _Float16*)d_in[0];
    const _Float16* lut = (const _Float16*)d_in[1];
    const int*      qw  = (const int*)d_in[2];
    float*          y   = (float*)d_out;
    (void)in_sizes; (void)n_in; (void)d_ws; (void)ws_size; (void)out_size;

    dim3 grid(BATCH / M_TILE, OUT_FEATURES / N_TILE);   // (16, 86); x-dim fastest => L2 reuse of W tile
    lutgemm_wmma_kernel<<<grid, dim3(256, 1, 1), 0, stream>>>(x, lut, qw, y);
}
